// LSTMDecoder_5205500363396
// MI455X (gfx1250) — matrix operand order repaired and checked by reading, not compiled, not run on hardware
//
#include <hip/hip_runtime.h>
#include <hip/hip_bf16.h>

// ---------------------------------------------------------------------------
// Problem constants: B=32, T=64, S=400, I=512, H=1024
// ---------------------------------------------------------------------------
#define BB   32
#define TT   64
#define SS   400
#define II   512
#define HH   1024
#define KG   (II + HH + HH)   // 2560 : K of gates GEMM  [x_t, ha, h]
#define NG   (4 * HH)         // 4096 : N of gates GEMM
#define KO   (2 * HH)         // 2048 : K of out GEMM    [c_enc, h]

typedef __attribute__((ext_vector_type(16))) __bf16 v16bf;
typedef __attribute__((ext_vector_type(8)))  float  v8f;

// d_out layout (flat f32, reference tuple order)
#define O_OUT   ((size_t)0)                    // output_ [B,T,H]
#define O_H     ((size_t)2097152)              // h_f
#define O_C     ((size_t)2129920)              // c_f
#define O_HA    ((size_t)2162688)              // ha_f
#define O_ATTN  ((size_t)2195456)              // attns [T,B,S]
#define O_PA    ((size_t)3014656)              // past_attn passthrough
#define O_PG    ((size_t)3027456)              // p_gen_out [B,T]
#define O_PD    ((size_t)3029504)              // past_dehy passthrough
#define O_LOSS  ((size_t)3029505)              // loss_cv (0)

static __device__ __forceinline__ unsigned short f32_to_bf16(float f) {
  unsigned int u = __float_as_uint(f);
  u += 0x7fffu + ((u >> 16) & 1u);   // round-to-nearest-even
  return (unsigned short)(u >> 16);
}
static __device__ __forceinline__ float sigmoidf_(float x) {
  return 1.0f / (1.0f + __expf(-x));
}

// ---------------------------------------------------------------------------
// WMMA GEMM:  C[M x N] = A_f32[M x K] * W_bf16[N x K]^T  (+ bias[N])
// Block: 256 threads = 8 waves; block tile 32(M) x 128(N); wave tile 16x32.
// K stepped by 64 per LDS stage -> 4 WMMAs per wave per stage.
// Double-buffered LDS: weight tiles stream global->LDS on the ASYNCcnt path
// (GLOBAL_LOAD_ASYNC_TO_LDS_B128) one full stage ahead of the WMMAs, so the
// async copy latency is hidden behind matrix work. Activation tiles are
// converted f32->bf16 through VGPRs (ds_store, DScnt-ordered by barriers).
// Requires: M % 32 == 0, N % 128 == 0, K % 64 == 0, 16B-aligned pointers.
// ---------------------------------------------------------------------------
#define LSTR 72   // LDS row stride in bf16 elems (144B -> keeps 16B alignment)

__global__ __launch_bounds__(256)
void gemm_wmma_bf16(const float* __restrict__ A,
                    const unsigned short* __restrict__ W,
                    const float* __restrict__ bias,
                    float* __restrict__ C,
                    int N, int K, int nTiles) {
  const int nb = blockIdx.x % nTiles;
  const int mb = blockIdx.x / nTiles;
  const int tid  = threadIdx.x;
  const int lane = tid & 31;
  const int wave = tid >> 5;
  const int mt = wave & 1;        // wave's M-tile (0..1)
  const int ng = wave >> 1;       // wave's 32-wide N group (0..3)
  const int half = lane >> 4;     // 0: lanes 0-15, 1: lanes 16-31
  const int l16  = lane & 15;
  const size_t mBase = (size_t)mb * 32;
  const size_t nBase = (size_t)nb * 128;

  __shared__ __align__(16) unsigned short As[2][32 * LSTR];
  __shared__ __align__(16) unsigned short Ws[2][128 * LSTR];

  v8f acc0 = {};
  v8f acc1 = {};

  // staging assignments
  const int arow = tid >> 3;           // 0..31
  const int acol = (tid & 7) * 8;      // 0,8,..,56
  const int wrow = tid >> 1;           // 0..127
  const int wcol = (tid & 1) * 32;     // 0 or 32

  const float* Ag = A + (mBase + arow) * (size_t)K + acol;
  const unsigned short* Wgp = W + (nBase + wrow) * (size_t)K + wcol;

  // stage K-chunk [k0, k0+64) into LDS buffer p
  auto stageA = [&](int k0, int p) {
    const float4 v0 = *(const float4*)(Ag + k0);
    const float4 v1 = *(const float4*)(Ag + k0 + 4);
    uint4 pk;
    pk.x = (unsigned)f32_to_bf16(v0.x) | ((unsigned)f32_to_bf16(v0.y) << 16);
    pk.y = (unsigned)f32_to_bf16(v0.z) | ((unsigned)f32_to_bf16(v0.w) << 16);
    pk.z = (unsigned)f32_to_bf16(v1.x) | ((unsigned)f32_to_bf16(v1.y) << 16);
    pk.w = (unsigned)f32_to_bf16(v1.z) | ((unsigned)f32_to_bf16(v1.w) << 16);
    *(uint4*)&As[p][arow * LSTR + acol] = pk;
  };
  auto stageW = [&](int k0, int p) {
    const unsigned short* g = Wgp + k0;
    // LDS byte offset = low 32 bits of the generic shared-memory address
    const unsigned ldsW = (unsigned)(size_t)&Ws[p][wrow * LSTR + wcol];
#pragma unroll
    for (int u = 0; u < 4; ++u) {
      asm volatile("global_load_async_to_lds_b128 %0, %1, off"
                   :: "v"(ldsW + (unsigned)(u * 16)), "v"(g + u * 8)
                   : "memory");
    }
  };

  const int nStage = K >> 6;           // K / 64
  stageA(0, 0);
  stageW(0, 0);

  for (int i = 0; i < nStage; ++i) {
    // stage i's async loads were issued one stage ago -> latency hidden
    asm volatile("s_wait_asynccnt 0" ::: "memory");
    __syncthreads();
    if (i + 1 < nStage) {              // prefetch next stage into other buffer
      stageA((i + 1) * 64, (i + 1) & 1);
      stageW((i + 1) * 64, (i + 1) & 1);
    }
    const unsigned short* Asp = As[i & 1];
    const unsigned short* Wsp = Ws[i & 1];
#pragma unroll
    for (int kc = 0; kc < 64; kc += 32) {
      union { uint4 u[2]; v16bf v; } fa, fb0, fb1;
      // A frag: elem e -> K = half*16 + e
      const unsigned short* ar = &Asp[(mt * 16 + l16) * LSTR + kc + half * 16];
      fa.u[0] = *(const uint4*)(ar);
      fa.u[1] = *(const uint4*)(ar + 8);
      // B frags: elem e -> K = half*16 + e
      const unsigned short* br0 = &Wsp[(ng * 32 + l16) * LSTR + kc + half * 16];
      fb0.u[0] = *(const uint4*)(br0);
      fb0.u[1] = *(const uint4*)(br0 + 8);
      const unsigned short* br1 = &Wsp[(ng * 32 + 16 + l16) * LSTR + kc + half * 16];
      fb1.u[0] = *(const uint4*)(br1);
      fb1.u[1] = *(const uint4*)(br1 + 8);

      acc0 = __builtin_amdgcn_wmma_f32_16x16x32_bf16(
                 false, fa.v, false, fb0.v, (short)0, acc0, false, false);
      acc1 = __builtin_amdgcn_wmma_f32_16x16x32_bf16(
                 false, fa.v, false, fb1.v, (short)0, acc1, false, false);
    }
    __syncthreads();
  }

  // --- store D: VGPR r, lane -> (M = mt*16 + half*8 + r, N = l16 [+16]) ---
  const size_t n0 = nBase + ng * 32 + l16;
  const float bv0 = bias ? bias[n0] : 0.0f;
  const float bv1 = bias ? bias[n0 + 16] : 0.0f;
#pragma unroll
  for (int r = 0; r < 8; ++r) {
    const size_t m = mBase + mt * 16 + half * 8 + r;
    C[m * (size_t)N + n0]      = acc0[r] + bv0;
    C[m * (size_t)N + n0 + 16] = acc1[r] + bv1;
  }
}

// ---------------------------------------------------------------------------
// Weight packing (once per launch)
// ---------------------------------------------------------------------------
__global__ void conv_wg_kernel(const float* __restrict__ W_ih,
                               const float* __restrict__ W_hh,
                               unsigned short* __restrict__ Wg) {
  size_t i = (size_t)blockIdx.x * blockDim.x + threadIdx.x;
  if (i >= (size_t)NG * KG) return;
  const int n = (int)(i / KG), k = (int)(i % KG);
  const float v = (k < II + HH) ? W_ih[(size_t)n * (II + HH) + k]
                                : W_hh[(size_t)n * HH + (k - II - HH)];
  Wg[i] = f32_to_bf16(v);
}

__global__ void conv_f2bf_kernel(const float* __restrict__ src,
                                 unsigned short* __restrict__ dst, size_t n) {
  size_t i = (size_t)blockIdx.x * blockDim.x + threadIdx.x;
  if (i < n) dst[i] = f32_to_bf16(src[i]);
}

__global__ void init_state_kernel(const float* __restrict__ h0,
                                  const float* __restrict__ c0,
                                  const float* __restrict__ ha0,
                                  float* __restrict__ h, float* __restrict__ c,
                                  float* __restrict__ ha) {
  int i = blockIdx.x * blockDim.x + threadIdx.x;
  if (i < BB * HH) { h[i] = h0[i]; c[i] = c0[i]; ha[i] = ha0[i]; }
}

// ---------------------------------------------------------------------------
// Per-step kernels
// ---------------------------------------------------------------------------
__global__ void pack_xcat_kernel(const float* __restrict__ input_,
                                 const float* __restrict__ ha,
                                 const float* __restrict__ h,
                                 float* __restrict__ xcat, int t) {
  int i = blockIdx.x * blockDim.x + threadIdx.x;
  if (i >= BB * KG) return;
  const int b = i / KG, k = i % KG;
  float v;
  if (k < II)            v = input_[((size_t)b * TT + t) * II + k];
  else if (k < II + HH)  v = ha[b * HH + (k - II)];
  else                   v = h[b * HH + (k - II - HH)];
  xcat[i] = v;
}

__global__ void lstm_elem_kernel(const float* __restrict__ gates,
                                 const float* __restrict__ b_ih,
                                 const float* __restrict__ b_hh,
                                 float* __restrict__ h, float* __restrict__ c) {
  int i = blockIdx.x * blockDim.x + threadIdx.x;
  if (i >= BB * HH) return;
  const int b = i >> 10, j = i & (HH - 1);
  const float* g = gates + (size_t)b * NG;
  const float gi = g[j]          + b_ih[j]          + b_hh[j];
  const float gf = g[HH + j]     + b_ih[HH + j]     + b_hh[HH + j];
  const float gg = g[2 * HH + j] + b_ih[2 * HH + j] + b_hh[2 * HH + j];
  const float go = g[3 * HH + j] + b_ih[3 * HH + j] + b_hh[3 * HH + j];
  const float cn = sigmoidf_(gf) * c[i] + sigmoidf_(gi) * tanhf(gg);
  c[i] = cn;
  h[i] = sigmoidf_(go) * tanhf(cn);
}

// logits[b,s] = sum_t enhy[b,s,t]*h[b,t]; softmax over s -> attn slice of d_out
__global__ __launch_bounds__(256)
void attn_softmax_kernel(const float* __restrict__ enhy,
                         const float* __restrict__ h,
                         float* __restrict__ attn_out) {
  const int b = blockIdx.x, tid = threadIdx.x;
  const int lane = tid & 31, wave = tid >> 5;
  __shared__ float hs[HH];
  __shared__ float lg[SS];
  __shared__ float red[256];
  for (int j = tid; j < HH; j += 256) hs[j] = h[b * HH + j];
  __syncthreads();
  for (int s = wave; s < SS; s += 8) {
    const float* row = enhy + ((size_t)b * SS + s) * HH;
    float p = 0.0f;
    for (int j = lane; j < HH; j += 32) p += row[j] * hs[j];
#pragma unroll
    for (int off = 16; off > 0; off >>= 1) p += __shfl_down(p, off, 32);
    if (lane == 0) lg[s] = p;
  }
  __syncthreads();
  float m = -3.4e38f;
  for (int s = tid; s < SS; s += 256) m = fmaxf(m, lg[s]);
  red[tid] = m; __syncthreads();
  for (int st = 128; st > 0; st >>= 1) {
    if (tid < st) red[tid] = fmaxf(red[tid], red[tid + st]);
    __syncthreads();
  }
  m = red[0]; __syncthreads();
  float sum = 0.0f;
  for (int s = tid; s < SS; s += 256) { float e = __expf(lg[s] - m); lg[s] = e; sum += e; }
  red[tid] = sum; __syncthreads();
  for (int st = 128; st > 0; st >>= 1) {
    if (tid < st) red[tid] += red[tid + st];
    __syncthreads();
  }
  const float inv = 1.0f / red[0];
  for (int s = tid; s < SS; s += 256) attn_out[b * SS + s] = lg[s] * inv;
}

// c_enc[b,j] = sum_s attn[b,s]*enc[b,s,j]; also pack cat2 = [c_enc, h] (f32)
__global__ __launch_bounds__(256)
void cenc_pack_kernel(const float* __restrict__ attn,
                      const float* __restrict__ enc,
                      const float* __restrict__ h,
                      float* __restrict__ cenc, float* __restrict__ cat2) {
  const int b = blockIdx.x, tid = threadIdx.x;
  __shared__ float at[SS];
  for (int s = tid; s < SS; s += 256) at[s] = attn[b * SS + s];
  __syncthreads();
  for (int j = tid; j < HH; j += 256) {
    const float* base = enc + (size_t)b * SS * HH + j;
    float acc = 0.0f;
    for (int s = 0; s < SS; ++s) acc += at[s] * base[(size_t)s * HH];
    cenc[b * HH + j] = acc;
    cat2[(size_t)b * KO + j] = acc;
    cat2[(size_t)b * KO + HH + j] = h[b * HH + j];
  }
}

// p_gen[b,t] = sigmoid([x_t,h,c_enc].W_pt + b_pt); also copy ha -> output_
__global__ __launch_bounds__(256)
void pt_out_kernel(const float* __restrict__ input_,
                   const float* __restrict__ h,
                   const float* __restrict__ cenc,
                   const float* __restrict__ ha,
                   const float* __restrict__ W_pt,
                   const float* __restrict__ b_pt,
                   float* __restrict__ out_output,
                   float* __restrict__ out_pgen, int t) {
  const int b = blockIdx.x, tid = threadIdx.x;
  for (int j = tid; j < HH; j += 256)
    out_output[((size_t)b * TT + t) * HH + j] = ha[b * HH + j];
  float p = 0.0f;
  for (int k = tid; k < II + 2 * HH; k += 256) {
    float v;
    if (k < II)           v = input_[((size_t)b * TT + t) * II + k];
    else if (k < II + HH) v = h[b * HH + (k - II)];
    else                  v = cenc[b * HH + (k - II - HH)];
    p += v * W_pt[k];
  }
  __shared__ float red[256];
  red[tid] = p; __syncthreads();
  for (int st = 128; st > 0; st >>= 1) {
    if (tid < st) red[tid] += red[tid + st];
    __syncthreads();
  }
  if (tid == 0) out_pgen[b * TT + t] = sigmoidf_(red[0] + b_pt[0]);
}

__global__ void finalize_kernel(const float* __restrict__ h,
                                const float* __restrict__ c,
                                const float* __restrict__ ha,
                                const float* __restrict__ past_attn,
                                const float* __restrict__ past_dehy,
                                float* __restrict__ out) {
  int i = blockIdx.x * blockDim.x + threadIdx.x;
  if (i < BB * HH) {
    out[O_H + i]  = h[i];
    out[O_C + i]  = c[i];
    out[O_HA + i] = ha[i];
  }
  if (i < BB * SS) out[O_PA + i] = past_attn[i];
  if (i == 0) { out[O_PD] = past_dehy[0]; out[O_LOSS] = 0.0f; }
}

// ---------------------------------------------------------------------------
// Launch
// ---------------------------------------------------------------------------
extern "C" void kernel_launch(void* const* d_in, const int* in_sizes, int n_in,
                              void* d_out, int out_size, void* d_ws, size_t ws_size,
                              hipStream_t stream) {
  (void)in_sizes; (void)n_in; (void)out_size; (void)ws_size;
  const float* input_     = (const float*)d_in[1];
  const float* h0         = (const float*)d_in[2];
  const float* c0         = (const float*)d_in[3];
  const float* h_attn0    = (const float*)d_in[4];
  const float* encoder_hy = (const float*)d_in[5];
  const float* past_attn  = (const float*)d_in[6];
  const float* past_dehy  = (const float*)d_in[8];
  const float* W_ih       = (const float*)d_in[9];
  const float* W_hh       = (const float*)d_in[10];
  const float* b_ih       = (const float*)d_in[11];
  const float* b_hh       = (const float*)d_in[12];
  const float* W_attn     = (const float*)d_in[13];
  const float* W_out      = (const float*)d_in[14];
  const float* b_out      = (const float*)d_in[15];
  const float* W_pt       = (const float*)d_in[16];
  const float* b_pt       = (const float*)d_in[17];
  float* out = (float*)d_out;

  // workspace layout (256B-aligned slabs)
  char* ws = (char*)d_ws;
  size_t off = 0;
  auto alloc = [&](size_t bytes) -> char* {
    char* p = ws + off;
    off += (bytes + 255) & ~(size_t)255;
    return p;
  };
  unsigned short* Wg    = (unsigned short*)alloc((size_t)NG * KG * 2);   // 20.97 MB
  unsigned short* Wout  = (unsigned short*)alloc((size_t)HH * KO * 2);   //  4.19 MB
  unsigned short* Wattn = (unsigned short*)alloc((size_t)HH * HH * 2);   //  2.10 MB
  float* enhy  = (float*)alloc((size_t)BB * SS * HH * 4);                // 52.43 MB
  float* xcat  = (float*)alloc((size_t)BB * KG * 4);
  float* cat2  = (float*)alloc((size_t)BB * KO * 4);
  float* gates = (float*)alloc((size_t)BB * NG * 4);
  float* hbuf  = (float*)alloc((size_t)BB * HH * 4);
  float* cbuf  = (float*)alloc((size_t)BB * HH * 4);
  float* habuf = (float*)alloc((size_t)BB * HH * 4);
  float* cenc  = (float*)alloc((size_t)BB * HH * 4);

  // --- hoisted work: weight packing + enhy = encoder_hy @ W_attn^T ---
  {
    const size_t nWg = (size_t)NG * KG;
    conv_wg_kernel<<<(unsigned)((nWg + 255) / 256), 256, 0, stream>>>(W_ih, W_hh, Wg);
    const size_t nWo = (size_t)HH * KO;
    conv_f2bf_kernel<<<(unsigned)((nWo + 255) / 256), 256, 0, stream>>>(W_out, Wout, nWo);
    const size_t nWa = (size_t)HH * HH;
    conv_f2bf_kernel<<<(unsigned)((nWa + 255) / 256), 256, 0, stream>>>(W_attn, Wattn, nWa);
    // M = B*S = 12800 (400 m-blocks), N = 1024 (8 n-tiles of 128), K = 1024
    gemm_wmma_bf16<<<400 * 8, 256, 0, stream>>>(encoder_hy, Wattn, nullptr,
                                                enhy, HH, HH, 8);
    init_state_kernel<<<(BB * HH + 255) / 256, 256, 0, stream>>>(
        h0, c0, h_attn0, hbuf, cbuf, habuf);
  }

  // --- recurrent loop ---
  for (int t = 0; t < TT; ++t) {
    pack_xcat_kernel<<<(BB * KG + 255) / 256, 256, 0, stream>>>(
        input_, habuf, hbuf, xcat, t);
    // gates[32 x 4096] = xcat[32 x 2560] . Wg^T
    gemm_wmma_bf16<<<NG / 128, 256, 0, stream>>>(xcat, Wg, nullptr, gates,
                                                 NG, KG, NG / 128);
    lstm_elem_kernel<<<(BB * HH + 255) / 256, 256, 0, stream>>>(
        gates, b_ih, b_hh, hbuf, cbuf);
    float* attn_t = out + O_ATTN + (size_t)t * BB * SS;   // [T,B,S] slice
    attn_softmax_kernel<<<BB, 256, 0, stream>>>(enhy, hbuf, attn_t);
    cenc_pack_kernel<<<BB, 256, 0, stream>>>(attn_t, encoder_hy, hbuf, cenc, cat2);
    // ha[32 x 1024] = cat2[32 x 2048] . Wout^T + b_out
    gemm_wmma_bf16<<<HH / 128, 256, 0, stream>>>(cat2, Wout, b_out, habuf,
                                                 HH, KO, HH / 128);
    pt_out_kernel<<<BB, 256, 0, stream>>>(input_, hbuf, cenc, habuf, W_pt, b_pt,
                                          out + O_OUT, out + O_PG, t);
  }

  finalize_kernel<<<(BB * HH + 255) / 256, 256, 0, stream>>>(
      hbuf, cbuf, habuf, past_attn, past_dehy, out);
}
